// Mamba_65429531788056
// MI455X (gfx1250) — matrix operand order repaired and checked by reading, not compiled, not run on hardware
//
#include <hip/hip_runtime.h>
#include <hip/hip_bf16.h>

// ---------------- problem constants ----------------
#define DM    768      // d_model
#define DS    16       // d_state
#define DCONV 4
#define DI    1536     // d_inner
#define RK    48       // dt_rank
#define KXR   80       // dt_rank + 2*d_state (rows of x_dbl)
#define XDP   96       // padded row stride for x_dblT
#define BSZ   2
#define LSEQ  2048
#define NCH   8        // scan chunks
#define CLEN  256      // LSEQ / NCH

typedef __attribute__((ext_vector_type(16))) __bf16 v16bf;
typedef __attribute__((ext_vector_type(8)))  float  v8f;

// ---------------- WMMA fragment helpers (CDNA5 16x16x32 bf16) ----------------
// A (16xK, M x K): lane m = lane&15; holds K = kb..kb+7 , kb+16..kb+23 (kb = lane>=16 ? 8 : 0)
// B (Kx16, K x N): lane n = lane&15; holds K = kb..kb+7 , kb+16..kb+23 (kb = lane>=16 ? 8 : 0)
// C/D (16x16 f32): VGPR v -> M = v + (lane>=16 ? 8 : 0), N = lane&15

__device__ __forceinline__ v8f wmma_bf16(v16bf a, v16bf b, v8f c) {
  return __builtin_amdgcn_wmma_f32_16x16x32_bf16(false, a, false, b, (short)0, c,
                                                 false, false);
}

__device__ __forceinline__ v16bf frag_a_f32(const float* __restrict__ A, int ld,
                                            int m0, int k0, int lane) {
  const float* p = A + (size_t)(m0 + (lane & 15)) * ld + k0 + ((lane & 16) ? 8 : 0);
  v16bf f;
#pragma unroll
  for (int i = 0; i < 8; ++i) { f[i] = (__bf16)p[i]; f[i + 8] = (__bf16)p[16 + i]; }
  return f;
}

// A fragment with zero-padding for k >= kmax (used for W_dt, K=48 padded to 64)
__device__ __forceinline__ v16bf frag_a_f32_kpad(const float* __restrict__ A, int ld,
                                                 int kmax, int m0, int k0, int lane) {
  int m  = m0 + (lane & 15);
  int kb = k0 + ((lane & 16) ? 8 : 0);
  v16bf f;
#pragma unroll
  for (int i = 0; i < 8; ++i) {
    int ka = kb + i, kc = kb + 16 + i;
    f[i]     = (ka < kmax) ? (__bf16)A[(size_t)m * ld + ka] : (__bf16)0.0f;
    f[i + 8] = (kc < kmax) ? (__bf16)A[(size_t)m * ld + kc] : (__bf16)0.0f;
  }
  return f;
}

// A fragment = 0.5*(Y0+Y1), bf16 sources (fused branch combine for out-proj)
__device__ __forceinline__ v16bf frag_a_avg(const __bf16* __restrict__ Y0,
                                            const __bf16* __restrict__ Y1, int ld,
                                            int m0, int k0, int lane) {
  size_t off = (size_t)(m0 + (lane & 15)) * ld + k0 + ((lane & 16) ? 8 : 0);
  const __bf16* p0 = Y0 + off;
  const __bf16* p1 = Y1 + off;
  v16bf f;
#pragma unroll
  for (int i = 0; i < 8; ++i) {
    f[i]     = (__bf16)(0.5f * ((float)p0[i] + (float)p1[i]));
    f[i + 8] = (__bf16)(0.5f * ((float)p0[16 + i] + (float)p1[16 + i]));
  }
  return f;
}

// B fragment from [N][K] f32 source (K contiguous)
__device__ __forceinline__ v16bf frag_b_f32(const float* __restrict__ Bs, int ld,
                                            int n0, int k0, int lane) {
  const float* p = Bs + (size_t)(n0 + (lane & 15)) * ld + k0 + ((lane & 16) ? 8 : 0);
  v16bf f;
#pragma unroll
  for (int i = 0; i < 8; ++i) { f[i] = (__bf16)p[i]; f[i + 8] = (__bf16)p[16 + i]; }
  return f;
}

// B fragment from [N][K] bf16 source
__device__ __forceinline__ v16bf frag_b_bf16(const __bf16* __restrict__ Bs, int ld,
                                             int n0, int k0, int lane) {
  const __bf16* p = Bs + (size_t)(n0 + (lane & 15)) * ld + k0 + ((lane & 16) ? 8 : 0);
  v16bf f;
#pragma unroll
  for (int i = 0; i < 8; ++i) { f[i] = p[i]; f[i + 8] = p[16 + i]; }
  return f;
}

__device__ __forceinline__ v8f zero8() {
  v8f z = {0.f, 0.f, 0.f, 0.f, 0.f, 0.f, 0.f, 0.f};
  return z;
}

// ---------------- Kernel 1: in-projection  xz[b,e,l] = sum_d W_in[e,d]*hs[b,l,d] ----
// Wave macro-tile: 16M x 64N (4 accumulators, 4 independent WMMAs per K-step).
// grid (L/64, 3072/16/8, B), block 256 (8 waves -> 8 adjacent M-tiles share B frags)
__global__ __launch_bounds__(256) void k_gemm_in(const float* __restrict__ hs,
                                                 const float* __restrict__ Win,
                                                 float* __restrict__ xz) {
  int lane = threadIdx.x & 31, wave = threadIdx.x >> 5;
  int ltile = blockIdx.x * 64;
  int mtile = (blockIdx.y * 8 + wave) * 16;
  int b = blockIdx.z;
  const float* Bsrc = hs + (size_t)b * LSEQ * DM;   // [l][d], d contiguous
  v8f acc0 = zero8(), acc1 = zero8(), acc2 = zero8(), acc3 = zero8();
  for (int k = 0; k < DM; k += 32) {
    v16bf a  = frag_a_f32(Win, DM, mtile, k, lane);
    v16bf b0 = frag_b_f32(Bsrc, DM, ltile +  0, k, lane);
    v16bf b1 = frag_b_f32(Bsrc, DM, ltile + 16, k, lane);
    v16bf b2 = frag_b_f32(Bsrc, DM, ltile + 32, k, lane);
    v16bf b3 = frag_b_f32(Bsrc, DM, ltile + 48, k, lane);
    acc0 = wmma_bf16(a, b0, acc0);
    acc1 = wmma_bf16(a, b1, acc1);
    acc2 = wmma_bf16(a, b2, acc2);
    acc3 = wmma_bf16(a, b3, acc3);
  }
  float* out = xz + (size_t)b * (2 * DI) * LSEQ;
  int mo = (lane & 16) ? 8 : 0;
  v8f accs[4] = {acc0, acc1, acc2, acc3};
#pragma unroll
  for (int t = 0; t < 4; ++t) {
    int l = ltile + t * 16 + (lane & 15);
#pragma unroll
    for (int v = 0; v < 8; ++v)
      out[(size_t)(mtile + mo + v) * LSEQ + l] = accs[t][v];
  }
}

// ---------------- Kernel 2: causal depthwise conv + SiLU, both branches --------
// br=1 operates on the time-reversed x. Output transposed: xcT[zid][l][d] (bf16)
// grid (L/64, DI/16, 2*BSZ), block 256
__global__ __launch_bounds__(256) void k_conv_silu(const float* __restrict__ xz,
                                                   const float* __restrict__ cw0,
                                                   const float* __restrict__ cb0,
                                                   const float* __restrict__ cw1,
                                                   const float* __restrict__ cb1,
                                                   __bf16* __restrict__ xcT) {
  int zid = blockIdx.z;            // br*BSZ + b
  int br = zid >> 1, b = zid & 1;
  const float* cw = br ? cw1 : cw0;
  const float* cb = br ? cb1 : cb0;
  int d0 = blockIdx.y * 16;
  int l0 = blockIdx.x * 64;
  const float* X = xz + (size_t)b * (2 * DI) * LSEQ;     // x half: rows 0..DI-1
  __bf16* out = xcT + (size_t)zid * LSEQ * DI;
  for (int idx = threadIdx.x; idx < 1024; idx += 256) {
    int li = idx & 63, di = idx >> 6;
    int d = d0 + di, l = l0 + li;
    const float* w = cw + (size_t)d * DCONV;
    float acc = cb[d];
#pragma unroll
    for (int j = 0; j < DCONV; ++j) {
      int lj = l - (DCONV - 1) + j;
      float xv = 0.f;
      if (lj >= 0) {
        int lp = br ? (LSEQ - 1 - lj) : lj;
        xv = X[(size_t)d * LSEQ + lp];
      }
      acc += w[j] * xv;
    }
    float s = acc / (1.f + __expf(-acc));                 // SiLU
    out[(size_t)l * DI + d] = (__bf16)s;
  }
}

// ---------------- Kernel 3: x_dbl = W_x @ x_conv ; stored transposed ------------
// Wave macro-tile 16M x 64N. xdT[zid][l][0..79] (f32, row stride 96).
// grid (L/64, 5, 2*BSZ), block 32
__global__ __launch_bounds__(32) void k_gemm_xdbl(const float* __restrict__ Wx0,
                                                  const float* __restrict__ Wx1,
                                                  const __bf16* __restrict__ xcT,
                                                  float* __restrict__ xdT) {
  int lane = threadIdx.x & 31;
  int zid = blockIdx.z;
  int br = zid >> 1;
  int mtile = blockIdx.y * 16;        // 5 tiles cover the 80 rows exactly
  int ltile = blockIdx.x * 64;
  const float* A = br ? Wx1 : Wx0;                        // [80][DI]
  const __bf16* Bs = xcT + (size_t)zid * LSEQ * DI;       // [l][DI]
  v8f acc0 = zero8(), acc1 = zero8(), acc2 = zero8(), acc3 = zero8();
  for (int k = 0; k < DI; k += 32) {
    v16bf a  = frag_a_f32(A, DI, mtile, k, lane);
    v16bf b0 = frag_b_bf16(Bs, DI, ltile +  0, k, lane);
    v16bf b1 = frag_b_bf16(Bs, DI, ltile + 16, k, lane);
    v16bf b2 = frag_b_bf16(Bs, DI, ltile + 32, k, lane);
    v16bf b3 = frag_b_bf16(Bs, DI, ltile + 48, k, lane);
    acc0 = wmma_bf16(a, b0, acc0);
    acc1 = wmma_bf16(a, b1, acc1);
    acc2 = wmma_bf16(a, b2, acc2);
    acc3 = wmma_bf16(a, b3, acc3);
  }
  int mo = (lane & 16) ? 8 : 0;
  v8f accs[4] = {acc0, acc1, acc2, acc3};
#pragma unroll
  for (int t = 0; t < 4; ++t) {
    float* out = xdT + ((size_t)zid * LSEQ + ltile + t * 16 + (lane & 15)) * XDP
               + mtile + mo;
#pragma unroll
    for (int v = 0; v < 8; ++v) out[v] = accs[t][v];      // M contiguous across VGPRs
  }
}

// ---------------- Kernel 4: dt = softplus(W_dt @ dt_lo + b_dt), transposed -----
// A is K-padded to 64 with zeros so B can read through the Bm region safely.
// Wave macro-tile 16M x 64N. dtT[zid][l][d].
// grid (L/64, DI/16/8, 2*BSZ), block 256
__global__ __launch_bounds__(256) void k_gemm_dt(const float* __restrict__ Wdt0,
                                                 const float* __restrict__ Wdt1,
                                                 const float* __restrict__ bdt0,
                                                 const float* __restrict__ bdt1,
                                                 const float* __restrict__ xdT,
                                                 float* __restrict__ dtT) {
  int lane = threadIdx.x & 31, wave = threadIdx.x >> 5;
  int zid = blockIdx.z;
  int br = zid >> 1;
  int mtile = (blockIdx.y * 8 + wave) * 16;
  int ltile = blockIdx.x * 64;
  const float* A  = br ? Wdt1 : Wdt0;                     // [DI][48]
  const float* bd = br ? bdt1 : bdt0;
  const float* Bs = xdT + (size_t)zid * LSEQ * XDP;       // [l][96], cols 0..63 used
  v8f acc0 = zero8(), acc1 = zero8(), acc2 = zero8(), acc3 = zero8();
#pragma unroll
  for (int k = 0; k < 64; k += 32) {
    v16bf a  = frag_a_f32_kpad(A, RK, RK, mtile, k, lane);
    v16bf b0 = frag_b_f32(Bs, XDP, ltile +  0, k, lane);
    v16bf b1 = frag_b_f32(Bs, XDP, ltile + 16, k, lane);
    v16bf b2 = frag_b_f32(Bs, XDP, ltile + 32, k, lane);
    v16bf b3 = frag_b_f32(Bs, XDP, ltile + 48, k, lane);
    acc0 = wmma_bf16(a, b0, acc0);
    acc1 = wmma_bf16(a, b1, acc1);
    acc2 = wmma_bf16(a, b2, acc2);
    acc3 = wmma_bf16(a, b3, acc3);
  }
  int mo = (lane & 16) ? 8 : 0;
  float bias[8];
#pragma unroll
  for (int v = 0; v < 8; ++v) bias[v] = bd[mtile + mo + v];
  v8f accs[4] = {acc0, acc1, acc2, acc3};
#pragma unroll
  for (int t = 0; t < 4; ++t) {
    float* out = dtT + ((size_t)zid * LSEQ + ltile + t * 16 + (lane & 15)) * DI
               + mtile + mo;
#pragma unroll
    for (int v = 0; v < 8; ++v) {
      float x = accs[t][v] + bias[v];
      out[v] = (x > 20.f) ? x : __logf(1.f + __expf(x));  // softplus
    }
  }
}

// ---------------- Kernel 5: scan pass 1 — per-chunk affine carries -------------
// Per (zid, chunk, d): carryA[n] = prod(dA), carryB[n] = affine accum.
// grid (DI/256, NCH, 2*BSZ), block 256
__global__ __launch_bounds__(256) void k_scan1(const float* __restrict__ Alog0,
                                               const float* __restrict__ Alog1,
                                               const float* __restrict__ dtT,
                                               const __bf16* __restrict__ xcT,
                                               const float* __restrict__ xdT,
                                               float* __restrict__ carry) {
  int zid = blockIdx.z;
  int br = zid >> 1;
  int d = blockIdx.x * 256 + threadIdx.x;
  int c = blockIdx.y;
  const float* Alog = br ? Alog1 : Alog0;
  float Ar[DS], cA[DS], cB[DS];
#pragma unroll
  for (int n = 0; n < DS; ++n) {
    Ar[n] = -__expf(Alog[(size_t)d * DS + n]);
    cA[n] = 1.f; cB[n] = 0.f;
  }
  const float*  dts = dtT + (size_t)zid * LSEQ * DI + d;
  const __bf16* xs  = xcT + (size_t)zid * LSEQ * DI + d;
  const float*  bc  = xdT + (size_t)zid * LSEQ * XDP;
  int l0 = c * CLEN;
  for (int l = l0; l < l0 + CLEN; ++l) {
    float dt  = dts[(size_t)l * DI];
    float xv  = (float)xs[(size_t)l * DI];
    float dtx = dt * xv;
    const float* Bt = bc + (size_t)l * XDP + RK;          // B_t (16 floats)
#pragma unroll
    for (int n = 0; n < DS; ++n) {
      float dA = __expf(dt * Ar[n]);
      cA[n] *= dA;
      cB[n] = dA * cB[n] + dtx * Bt[n];
    }
  }
  float* out = carry + (((size_t)zid * NCH + c) * DI + d) * (2 * DS);
#pragma unroll
  for (int n = 0; n < DS; ++n) { out[n] = cA[n]; out[DS + n] = cB[n]; }
}

// ---------------- Kernel 6: scan pass 2 — replay with carries, gate, un-reverse -
// grid (DI/256, NCH, 2*BSZ), block 256
__global__ __launch_bounds__(256) void k_scan2(const float* __restrict__ Alog0,
                                               const float* __restrict__ Alog1,
                                               const float* __restrict__ Dp0,
                                               const float* __restrict__ Dp1,
                                               const float* __restrict__ dtT,
                                               const __bf16* __restrict__ xcT,
                                               const float* __restrict__ xdT,
                                               const float* __restrict__ carry,
                                               const float* __restrict__ xz,
                                               __bf16* __restrict__ yg) {
  int zid = blockIdx.z;
  int br = zid >> 1, b = zid & 1;
  int d = blockIdx.x * 256 + threadIdx.x;
  int c = blockIdx.y;
  const float* Alog = br ? Alog1 : Alog0;
  float Dv = br ? Dp1[d] : Dp0[d];
  float Ar[DS], h[DS];
#pragma unroll
  for (int n = 0; n < DS; ++n) {
    Ar[n] = -__expf(Alog[(size_t)d * DS + n]);
    h[n] = 0.f;
  }
  // compose carries of chunks 0..c-1 -> h at chunk start
  for (int j = 0; j < c; ++j) {
    const float* cj = carry + (((size_t)zid * NCH + j) * DI + d) * (2 * DS);
#pragma unroll
    for (int n = 0; n < DS; ++n) h[n] = cj[n] * h[n] + cj[DS + n];
  }
  const float*  dts = dtT + (size_t)zid * LSEQ * DI + d;
  const __bf16* xs  = xcT + (size_t)zid * LSEQ * DI + d;
  const float*  bc  = xdT + (size_t)zid * LSEQ * XDP;
  const float*  Z   = xz + (size_t)b * (2 * DI) * LSEQ + (size_t)(DI + d) * LSEQ;
  __bf16* out = yg + (size_t)zid * LSEQ * DI + d;
  int l0 = c * CLEN;
  for (int l = l0; l < l0 + CLEN; ++l) {
    float dt  = dts[(size_t)l * DI];
    float xv  = (float)xs[(size_t)l * DI];
    float dtx = dt * xv;
    const float* Bt = bc + (size_t)l * XDP + RK;
    const float* Ct = Bt + DS;
    float y = 0.f;
#pragma unroll
    for (int n = 0; n < DS; ++n) {
      float dA = __expf(dt * Ar[n]);
      h[n] = dA * h[n] + dtx * Bt[n];
      y += h[n] * Ct[n];
    }
    y += Dv * xv;
    int lp = br ? (LSEQ - 1 - l) : l;                     // un-reverse branch 1
    float zv = Z[lp];
    y *= zv / (1.f + __expf(-zv));                        // * SiLU(z)
    out[(size_t)lp * DI] = (__bf16)y;
  }
}

// ---------------- Kernel 7: out-projection, fused 0.5*(y + y_b) ----------------
// out[b,l,m] = sum_k 0.5*(yg0+yg1)[b,l,k] * W_out[m,k]
// Wave macro-tile 16M(l) x 64N(m).  grid (DM/64, (L/16)/8, B), block 256
__global__ __launch_bounds__(256) void k_gemm_out(const __bf16* __restrict__ yg,
                                                  const float* __restrict__ Wout,
                                                  float* __restrict__ out) {
  int lane = threadIdx.x & 31, wave = threadIdx.x >> 5;
  int b = blockIdx.z;
  int ltile = (blockIdx.y * 8 + wave) * 16;               // M = l
  int ntile = blockIdx.x * 64;                            // N = m
  const __bf16* Y0 = yg + (size_t)(0 * BSZ + b) * LSEQ * DI;
  const __bf16* Y1 = yg + (size_t)(1 * BSZ + b) * LSEQ * DI;
  v8f acc0 = zero8(), acc1 = zero8(), acc2 = zero8(), acc3 = zero8();
  for (int k = 0; k < DI; k += 32) {
    v16bf a  = frag_a_avg(Y0, Y1, DI, ltile, k, lane);
    v16bf b0 = frag_b_f32(Wout, DI, ntile +  0, k, lane);
    v16bf b1 = frag_b_f32(Wout, DI, ntile + 16, k, lane);
    v16bf b2 = frag_b_f32(Wout, DI, ntile + 32, k, lane);
    v16bf b3 = frag_b_f32(Wout, DI, ntile + 48, k, lane);
    acc0 = wmma_bf16(a, b0, acc0);
    acc1 = wmma_bf16(a, b1, acc1);
    acc2 = wmma_bf16(a, b2, acc2);
    acc3 = wmma_bf16(a, b3, acc3);
  }
  float* o = out + (size_t)b * LSEQ * DM;
  int mo = (lane & 16) ? 8 : 0;
  v8f accs[4] = {acc0, acc1, acc2, acc3};
#pragma unroll
  for (int t = 0; t < 4; ++t) {
    int n = ntile + t * 16 + (lane & 15);
#pragma unroll
    for (int v = 0; v < 8; ++v)
      o[(size_t)(ltile + mo + v) * DM + n] = accs[t][v];
  }
}

// ---------------- host launcher ------------------------------------------------
extern "C" void kernel_launch(void* const* d_in, const int* in_sizes, int n_in,
                              void* d_out, int out_size, void* d_ws, size_t ws_size,
                              hipStream_t stream) {
  const float* hs      = (const float*)d_in[0];
  const float* W_in    = (const float*)d_in[1];
  const float* conv_w  = (const float*)d_in[2];
  const float* conv_b  = (const float*)d_in[3];
  const float* conv_wb = (const float*)d_in[4];
  const float* conv_bb = (const float*)d_in[5];
  const float* W_x     = (const float*)d_in[6];
  const float* W_x_b   = (const float*)d_in[7];
  const float* W_dt    = (const float*)d_in[8];
  const float* b_dt    = (const float*)d_in[9];
  const float* W_dt_b  = (const float*)d_in[10];
  const float* b_dt_b  = (const float*)d_in[11];
  const float* A_log   = (const float*)d_in[12];
  const float* A_b_log = (const float*)d_in[13];
  const float* Dp      = (const float*)d_in[14];
  const float* Dp_b    = (const float*)d_in[15];
  const float* W_out   = (const float*)d_in[16];
  float* out = (float*)d_out;

  // workspace carve-up (bytes)
  char* ws = (char*)d_ws;
  size_t o_xz = 0;                                              // f32 [B][3072][L]
  size_t o_xc = o_xz + (size_t)BSZ * 2 * DI * LSEQ * 4;         // bf16 [4][L][DI]
  size_t o_xd = o_xc + (size_t)4 * LSEQ * DI * 2;               // f32 [4][L][96]
  size_t o_dt = o_xd + (size_t)4 * LSEQ * XDP * 4;              // f32 [4][L][DI]
  size_t o_cr = o_dt + (size_t)4 * LSEQ * DI * 4;               // f32 [4][NCH][DI][32]
  size_t o_yg = o_cr + (size_t)4 * NCH * DI * 2 * DS * 4;       // bf16 [4][L][DI]

  float*  xz    = (float*)(ws + o_xz);
  __bf16* xcT   = (__bf16*)(ws + o_xc);
  float*  xdT   = (float*)(ws + o_xd);
  float*  dtT   = (float*)(ws + o_dt);
  float*  carry = (float*)(ws + o_cr);
  __bf16* yg    = (__bf16*)(ws + o_yg);

  // 1) in-projection (WMMA bf16, 16x64 macro-tiles)
  k_gemm_in<<<dim3(LSEQ / 64, (2 * DI) / 16 / 8, BSZ), 256, 0, stream>>>(hs, W_in, xz);

  // 2) depthwise causal conv + SiLU (fwd + reversed branches), transposed output
  k_conv_silu<<<dim3(LSEQ / 64, DI / 16, 2 * BSZ), 256, 0, stream>>>(
      xz, conv_w, conv_b, conv_wb, conv_bb, xcT);

  // 3) x_dbl projection (WMMA bf16), transposed output
  k_gemm_xdbl<<<dim3(LSEQ / 64, KXR / 16, 2 * BSZ), 32, 0, stream>>>(W_x, W_x_b, xcT, xdT);

  // 4) dt projection + bias + softplus (WMMA bf16, zero-padded K)
  k_gemm_dt<<<dim3(LSEQ / 64, DI / 16 / 8, 2 * BSZ), 256, 0, stream>>>(
      W_dt, W_dt_b, b_dt, b_dt_b, xdT, dtT);

  // 5) chunked selective scan, pass 1: per-chunk affine carries
  k_scan1<<<dim3(DI / 256, NCH, 2 * BSZ), 256, 0, stream>>>(
      A_log, A_b_log, dtT, xcT, xdT, carry);

  // 6) pass 2: replay with composed carries, +D*x, *SiLU(z), un-reverse bwd branch
  k_scan2<<<dim3(DI / 256, NCH, 2 * BSZ), 256, 0, stream>>>(
      A_log, A_b_log, Dp, Dp_b, dtT, xcT, xdT, carry, xz, yg);

  // 7) out-projection with fused 0.5*(y + y_b) (WMMA bf16)
  k_gemm_out<<<dim3(DM / 64, LSEQ / 16 / 8, BSZ), 256, 0, stream>>>(yg, W_out, out);

  (void)in_sizes; (void)n_in; (void)out_size; (void)ws_size;
}
